// MambaLayer_45655502356849
// MI455X (gfx1250) — matrix operand order repaired and checked by reading, not compiled, not run on hardware
//
#include <hip/hip_runtime.h>
#include <cstddef>

// ---------------------------------------------------------------------------
// Model constants (from reference)
// ---------------------------------------------------------------------------
#define BB      2
#define LL      2048
#define MM      (BB * LL)        // 4096 rows
#define DMODEL  1024
#define NHEAD   16
#define HDATT   64
#define DINNER  2048
#define DSTATE  128
#define HEADDIM 64
#define HSSM    32
#define DCONV   4
#define CHUNK   256
#define NCHUNK  (LL / CHUNK)     // 8
#define CONVDIM (DINNER + 2 * DSTATE)       // 2304
#define DINPROJ (2 * DINNER + 2 * DSTATE + HSSM) // 4384
#define DFF     2048

typedef __attribute__((ext_vector_type(8)))  float  v8f;
typedef __attribute__((ext_vector_type(16))) __bf16 v16bf;
typedef __attribute__((ext_vector_type(8)))  __bf16 v8bf;

// ---------------------------------------------------------------------------
// WMMA helpers (CDNA5: v_wmma_f32_16x16x32_bf16, wave32)
// ---------------------------------------------------------------------------
__device__ inline v8f wmma_bf16(v16bf a, v16bf b, v8f c) {
  return __builtin_amdgcn_wmma_f32_16x16x32_bf16(false, a, false, b, (short)0, c,
                                                 false, false);
}

__device__ inline v8f zero8() {
  v8f z;
#pragma unroll
  for (int i = 0; i < 8; ++i) z[i] = 0.f;
  return z;
}

// Branch-free vector stage: 8 fp32 -> 8 bf16 -> one 16B LDS store.
// src must be 32B aligned, dst 16B aligned.
__device__ inline void stage8(__bf16* dst, const float* src, float msk) {
  const float4* s4 = reinterpret_cast<const float4*>(src);
  float4 x = s4[0];
  float4 y = s4[1];
  v8bf t;
  t[0] = (__bf16)(x.x * msk); t[1] = (__bf16)(x.y * msk);
  t[2] = (__bf16)(x.z * msk); t[3] = (__bf16)(x.w * msk);
  t[4] = (__bf16)(y.x * msk); t[5] = (__bf16)(y.y * msk);
  t[6] = (__bf16)(y.z * msk); t[7] = (__bf16)(y.w * msk);
  *reinterpret_cast<v8bf*>(dst) = t;
}

// A-fragment (16x32 MxK) from row-major fp32, optional scale.
// Layout used here (contiguous k): lane m = lane&15, elem i -> k = half*16+i
__device__ inline v16bf load_a_g(const float* base, int ld, int lane, float scale) {
  int m = lane & 15, half = lane >> 4;
  const float* r = base + (size_t)m * ld;
  v16bf a;
#pragma unroll
  for (int i = 0; i < 8; ++i) a[i] = (__bf16)(r[half * 16 + i] * scale);
#pragma unroll
  for (int i = 0; i < 8; ++i) a[8 + i] = (__bf16)(r[half * 16 + 8 + i] * scale);
  return a;
}

// B-fragment (32x16 KxN) where Bmat[k][n] = W[n][k], W row-major (N x K).
__device__ inline v16bf load_bT_g(const float* base, int ld, int lane) {
  int n = lane & 15, half = lane >> 4;
  const float* r = base + (size_t)n * ld + half * 16;
  v16bf b;
#pragma unroll
  for (int i = 0; i < 16; ++i) b[i] = (__bf16)r[i];
  return b;
}

// B-fragment (32x16 KxN) where Bmat[k][n] = X[k][n], X row-major (K x N).
__device__ inline v16bf load_b_g(const float* base, int ld, int lane) {
  int n = lane & 15, half = lane >> 4;
  const float* r = base + (size_t)(half * 16) * ld + n;
  v16bf b;
#pragma unroll
  for (int i = 0; i < 16; ++i) b[i] = (__bf16)r[(size_t)i * ld];
  return b;
}

// Same loaders from LDS bf16
__device__ inline v16bf lds_a_bf(const __bf16* base, int ld, int lane) {
  int m = lane & 15, half = lane >> 4;
  const __bf16* r = base + m * ld;
  v16bf a;
#pragma unroll
  for (int i = 0; i < 8; ++i) a[i] = r[half * 16 + i];
#pragma unroll
  for (int i = 0; i < 8; ++i) a[8 + i] = r[half * 16 + 8 + i];
  return a;
}

__device__ inline v16bf lds_bT_bf(const __bf16* base, int ld, int lane) {
  int n = lane & 15, half = lane >> 4;
  const __bf16* r = base + n * ld + half * 16;
  v16bf b;
#pragma unroll
  for (int i = 0; i < 16; ++i) b[i] = r[i];
  return b;
}

__device__ inline v16bf lds_b_bf(const __bf16* base, int ld, int lane) {
  int n = lane & 15, half = lane >> 4;
  const __bf16* r = base + (half * 16) * ld + n;
  v16bf b;
#pragma unroll
  for (int i = 0; i < 16; ++i) b[i] = r[i * ld];
  return b;
}

// ---------------------------------------------------------------------------
// GEMM: C[M,N] = A[M,K] @ W[N,K]^T (+bias). Block tile 128x64, BK=64,
// 256 threads = 8 waves, each wave a 32x32 sub-tile (8 WMMAs per barrier).
// M is always a multiple of 128 here; N may have a ragged edge (4384).
// ---------------------------------------------------------------------------
__global__ __launch_bounds__(256) void gemm_xwt(const float* __restrict__ A,
                                                const float* __restrict__ W,
                                                const float* __restrict__ bias,
                                                float* __restrict__ C,
                                                int M, int N, int K) {
  __shared__ __bf16 As[128][72];
  __shared__ __bf16 Ws[64][72];
  int tid = threadIdx.x, lane = tid & 31, wave = tid >> 5;
  int wm = (wave >> 1) * 32;   // 0,32,64,96
  int wn = (wave & 1) * 32;    // 0,32
  size_t bm = (size_t)blockIdx.x * 128;
  size_t bn = (size_t)blockIdx.y * 64;
  v8f acc00 = zero8(), acc01 = zero8(), acc10 = zero8(), acc11 = zero8();
  int srow = tid >> 3;         // 0..31
  int scol = (tid & 7) * 8;    // 0..56
  for (int k0 = 0; k0 < K; k0 += 64) {
#pragma unroll
    for (int pp = 0; pp < 4; ++pp) {
      int row = srow + pp * 32;
      const float* src = A + (bm + row) * (size_t)K + k0 + scol;
      if (k0 + 64 < K) __builtin_prefetch(src + 64, 0, 1);
      stage8(&As[row][scol], src, 1.f);
    }
#pragma unroll
    for (int pp = 0; pp < 2; ++pp) {
      int row = srow + pp * 32;
      size_t gn = bn + row;
      float msk = gn < (size_t)N ? 1.f : 0.f;
      size_t gcl = gn < (size_t)N ? gn : (size_t)(N - 1);
      const float* src = W + gcl * (size_t)K + k0 + scol;
      if (k0 + 64 < K) __builtin_prefetch(src + 64, 0, 1);
      stage8(&Ws[row][scol], src, msk);
    }
    __syncthreads();
#pragma unroll
    for (int kk = 0; kk < 2; ++kk) {
      v16bf a0 = lds_a_bf(&As[wm][kk * 32], 72, lane);
      v16bf a1 = lds_a_bf(&As[wm + 16][kk * 32], 72, lane);
      v16bf b0 = lds_bT_bf(&Ws[wn][kk * 32], 72, lane);
      v16bf b1 = lds_bT_bf(&Ws[wn + 16][kk * 32], 72, lane);
      acc00 = wmma_bf16(a0, b0, acc00);
      acc01 = wmma_bf16(a0, b1, acc01);
      acc10 = wmma_bf16(a1, b0, acc10);
      acc11 = wmma_bf16(a1, b1, acc11);
    }
    __syncthreads();
  }
  int half = lane >> 4, nn = lane & 15;
#pragma unroll
  for (int mi = 0; mi < 2; ++mi) {
#pragma unroll
    for (int ni = 0; ni < 2; ++ni) {
      v8f acc = (mi == 0) ? (ni == 0 ? acc00 : acc01) : (ni == 0 ? acc10 : acc11);
      size_t gcol = bn + wn + ni * 16 + nn;
      float bv = (bias && gcol < (size_t)N) ? bias[gcol] : 0.f;
#pragma unroll
      for (int j = 0; j < 8; ++j) {
        size_t grow = bm + wm + mi * 16 + half * 8 + j;
        if (gcol < (size_t)N)
          C[grow * (size_t)N + gcol] = acc[j] + bv;
      }
    }
  }
}

// ---------------------------------------------------------------------------
// rmsnorm(a + alpha*b) * w  over width 1024 (b may be null)
// ---------------------------------------------------------------------------
__global__ __launch_bounds__(256) void addnorm(const float* __restrict__ a,
                                               const float* __restrict__ b,
                                               float alpha,
                                               const float* __restrict__ w,
                                               float* __restrict__ out) {
  int row = blockIdx.x;
  const float* ar = a + (size_t)row * DMODEL;
  const float* br = b ? b + (size_t)row * DMODEL : nullptr;
  float tv[4];
  float ss = 0.f;
#pragma unroll
  for (int i = 0; i < 4; ++i) {
    int cc = threadIdx.x + (i << 8);
    float t = ar[cc] + (br ? alpha * br[cc] : 0.f);
    tv[i] = t;
    ss += t * t;
  }
#pragma unroll
  for (int off = 16; off; off >>= 1) ss += __shfl_xor(ss, off, 32);
  __shared__ float red[8];
  if ((threadIdx.x & 31) == 0) red[threadIdx.x >> 5] = ss;
  __syncthreads();
  float tot = 0.f;
#pragma unroll
  for (int i = 0; i < 8; ++i) tot += red[i];
  float r = rsqrtf(tot / (float)DMODEL + 1e-5f);
  float* orow = out + (size_t)row * DMODEL;
#pragma unroll
  for (int i = 0; i < 4; ++i) {
    int cc = threadIdx.x + (i << 8);
    orow[cc] = tv[i] * r * w[cc];
  }
}

// ---------------------------------------------------------------------------
// Attention (non-causal), flash-style. One wave per (b,h, 16 q rows).
// qkv layout: [b, l, 3, 16, 64] -> row of 3072 (q@0, k@1024, v@2048).
// ---------------------------------------------------------------------------
__global__ __launch_bounds__(32) void attn_fwd(const float* __restrict__ qkv,
                                               float* __restrict__ O) {
  int qt = blockIdx.x;         // 0..127
  int bh = blockIdx.y;         // 0..31
  int b = bh >> 4, h = bh & 15;
  int lane = threadIdx.x;
  int half = lane >> 4, nn = lane & 15;
  const float scale = 0.125f;  // 1/sqrt(64)
  const size_t RS = 3 * DMODEL;
  const float* qbase = qkv + ((size_t)(b * LL + qt * 16)) * RS + h * 64;
  const float* kbase = qkv + (size_t)b * LL * RS + DMODEL + h * 64;
  const float* vbase = qkv + (size_t)b * LL * RS + 2 * DMODEL + h * 64;
  v16bf qa0 = load_a_g(qbase, RS, lane, scale);
  v16bf qa1 = load_a_g(qbase + 32, RS, lane, scale);
  float m_i[8], l_i[8];
#pragma unroll
  for (int j = 0; j < 8; ++j) { m_i[j] = -1e30f; l_i[j] = 0.f; }
  v8f o0 = zero8(), o1 = zero8(), o2 = zero8(), o3 = zero8();
  __shared__ __bf16 Pbuf[16][34];
  for (int kt = 0; kt < LL; kt += 32) {
    const float* kp0 = kbase + (size_t)kt * RS;
    const float* kp1 = kp0 + 16 * RS;
    v8f s0 = zero8(), s1 = zero8();
    s0 = wmma_bf16(qa0, load_bT_g(kp0, RS, lane), s0);
    s0 = wmma_bf16(qa1, load_bT_g(kp0 + 32, RS, lane), s0);
    s1 = wmma_bf16(qa0, load_bT_g(kp1, RS, lane), s1);
    s1 = wmma_bf16(qa1, load_bT_g(kp1 + 32, RS, lane), s1);
#pragma unroll
    for (int j = 0; j < 8; ++j) {
      float mx = fmaxf(s0[j], s1[j]);
#pragma unroll
      for (int off = 1; off < 16; off <<= 1) mx = fmaxf(mx, __shfl_xor(mx, off, 32));
      float mnew = fmaxf(m_i[j], mx);
      float corr = __expf(m_i[j] - mnew);
      float p0 = __expf(s0[j] - mnew);
      float p1 = __expf(s1[j] - mnew);
      float ls = p0 + p1;
#pragma unroll
      for (int off = 1; off < 16; off <<= 1) ls += __shfl_xor(ls, off, 32);
      l_i[j] = l_i[j] * corr + ls;
      m_i[j] = mnew;
      o0[j] *= corr; o1[j] *= corr; o2[j] *= corr; o3[j] *= corr;
      Pbuf[half * 8 + j][nn] = (__bf16)p0;
      Pbuf[half * 8 + j][16 + nn] = (__bf16)p1;
    }
    __syncthreads();
    v16bf pf = lds_a_bf(&Pbuf[0][0], 34, lane);
    const float* vp = vbase + (size_t)kt * RS;
    o0 = wmma_bf16(pf, load_b_g(vp + 0, RS, lane), o0);
    o1 = wmma_bf16(pf, load_b_g(vp + 16, RS, lane), o1);
    o2 = wmma_bf16(pf, load_b_g(vp + 32, RS, lane), o2);
    o3 = wmma_bf16(pf, load_b_g(vp + 48, RS, lane), o3);
    __syncthreads();
  }
#pragma unroll
  for (int j = 0; j < 8; ++j) {
    float inv = 1.f / l_i[j];
    size_t row = (size_t)b * LL + qt * 16 + half * 8 + j;
    float* op = O + row * DMODEL + h * 64;
    op[0 * 16 + nn] = o0[j] * inv;
    op[1 * 16 + nn] = o1[j] * inv;
    op[2 * 16 + nn] = o2[j] * inv;
    op[3 * 16 + nn] = o3[j] * inv;
  }
}

// ---------------------------------------------------------------------------
// Depthwise causal conv (width 4) + bias + silu on xBC slice of zxbcdt.
// ---------------------------------------------------------------------------
__global__ __launch_bounds__(256) void convsilu(const float* __restrict__ zx,
                                                const float* __restrict__ cw,
                                                const float* __restrict__ cb,
                                                float* __restrict__ xbc) {
  int idx = blockIdx.x * 256 + threadIdx.x;
  const int total = MM * CONVDIM;
  if (idx >= total) return;
  int c = idx % CONVDIM;
  int row = idx / CONVDIM;     // b*2048 + l
  int l = row & (LL - 1);
  float acc = cb[c];
#pragma unroll
  for (int i = 0; i < DCONV; ++i) {
    int li = l - 3 + i;
    if (li >= 0)
      acc += zx[(size_t)(row - 3 + i) * DINPROJ + DINNER + c] * cw[c * DCONV + i];
  }
  acc = acc / (1.f + __expf(-acc));
  xbc[(size_t)row * CONVDIM + c] = acc;
}

// ---------------------------------------------------------------------------
// dt = softplus(dt_raw + dt_bias); a = -exp(A_log)*dt; chunk cumsum of a.
// ---------------------------------------------------------------------------
__global__ __launch_bounds__(256) void dtprep(const float* __restrict__ zx,
                                              const float* __restrict__ dt_bias,
                                              const float* __restrict__ A_log,
                                              float* __restrict__ dtbuf,
                                              float* __restrict__ Acs,
                                              float* __restrict__ Alast) {
  int t = blockIdx.x * 256 + threadIdx.x;
  if (t >= BB * HSSM * NCHUNK) return;
  int c = t & 7;
  int h = (t >> 3) & 31;
  int b = t >> 8;
  float Aneg = -__expf(A_log[h]);
  float run = 0.f;
  float* acsrow = Acs + ((size_t)((b * HSSM + h) * NCHUNK) + c) * CHUNK;
  for (int s = 0; s < CHUNK; ++s) {
    int l = c * CHUNK + s;
    float d = zx[(size_t)(b * LL + l) * DINPROJ + (DINNER + CONVDIM) + h] + dt_bias[h];
    float sp = (d > 20.f) ? d : log1pf(__expf(d));
    dtbuf[(size_t)(b * LL + l) * HSSM + h] = sp;
    run += Aneg * sp;
    acsrow[s] = run;
  }
  Alast[(b * HSSM + h) * NCHUNK + c] = run;
}

// ---------------------------------------------------------------------------
// SSD intra-chunk: Y_diag and chunk states. Block per (b, chunk, head),
// 512 threads = 16 waves (one wave per 16-row l tile).
// xbc: [M, 2304] (xs 0..2047, B 2048..2175, C 2176..2303)
// ---------------------------------------------------------------------------
__global__ __launch_bounds__(512) void ssd_local(const float* __restrict__ xbc,
                                                 const float* __restrict__ dtbuf,
                                                 const float* __restrict__ Acs,
                                                 const float* __restrict__ Alast,
                                                 float* __restrict__ Ybuf,
                                                 float* __restrict__ states) {
  int blk = blockIdx.x;
  int h = blk & 31;
  int c = (blk >> 5) & 7;
  int b = blk >> 8;
  int tid = threadIdx.x, lane = tid & 31, wave = tid >> 5;
  int half = lane >> 4, nn = lane & 15;
  __shared__ __bf16 xdt[CHUNK][72];
  __shared__ float acs_s[CHUNK];
  __shared__ float dst_s[CHUNK];
  __shared__ __bf16 Pb[16][16][34];
  size_t rowbase = (size_t)b * LL + c * CHUNK;
  float alast = Alast[(b * HSSM + h) * NCHUNK + c];
  const float* acsrow = Acs + ((size_t)((b * HSSM + h) * NCHUNK) + c) * CHUNK;
  for (int i = tid; i < CHUNK; i += 512) {
    float av = acsrow[i];
    acs_s[i] = av;
    dst_s[i] = __expf(alast - av);
  }
  // vectorized xdt staging: 8 elems per store
  for (int i = tid; i < CHUNK * HEADDIM / 8; i += 512) {
    int s = i >> 3;
    int p = (i & 7) * 8;
    const float* src = xbc + (rowbase + s) * CONVDIM + h * 64 + p;
    float dv = dtbuf[(rowbase + s) * HSSM + h];
    stage8(&xdt[s][p], src, dv);
  }
  __syncthreads();

  // ---- Y_diag: wave handles l tile `wave` ----
  int lt = wave;
  const float* Cbase = xbc + (rowbase + lt * 16) * CONVDIM + DINNER + DSTATE;
  v16bf ca[4];
#pragma unroll
  for (int kk = 0; kk < 4; ++kk)
    ca[kk] = load_a_g(Cbase + kk * 32, CONVDIM, lane, 1.f);
  v8f y[4];
#pragma unroll
  for (int ni = 0; ni < 4; ++ni) y[ni] = zero8();
  for (int sp = 0; sp <= lt; sp += 2) {
    const float* Bb0 = xbc + (rowbase + sp * 16) * CONVDIM + DINNER;
    v8f g0 = zero8(), g1 = zero8();
#pragma unroll
    for (int kk = 0; kk < 4; ++kk)
      g0 = wmma_bf16(ca[kk], load_bT_g(Bb0 + kk * 32, CONVDIM, lane), g0);
    bool has1 = (sp + 1 <= lt);
    if (has1) {
      const float* Bb1 = Bb0 + 16 * CONVDIM;
#pragma unroll
      for (int kk = 0; kk < 4; ++kk)
        g1 = wmma_bf16(ca[kk], load_bT_g(Bb1 + kk * 32, CONVDIM, lane), g1);
    }
#pragma unroll
    for (int j = 0; j < 8; ++j) {
      int lrow = lt * 16 + half * 8 + j;
      float al = acs_s[lrow];
      int s0 = sp * 16 + nn;
      float v0 = (s0 <= lrow) ? g0[j] * __expf(al - acs_s[s0]) : 0.f;
      Pb[wave][half * 8 + j][nn] = (__bf16)v0;
      int s1 = sp * 16 + 16 + nn;
      float v1 = (has1 && s1 <= lrow) ? g1[j] * __expf(al - acs_s[s1]) : 0.f;
      Pb[wave][half * 8 + j][16 + nn] = (__bf16)v1;
    }
    v16bf pf = lds_a_bf(&Pb[wave][0][0], 34, lane);
#pragma unroll
    for (int ni = 0; ni < 4; ++ni)
      y[ni] = wmma_bf16(pf, lds_b_bf(&xdt[sp * 16][ni * 16], 72, lane), y[ni]);
  }
#pragma unroll
  for (int j = 0; j < 8; ++j) {
    size_t orow = (rowbase + lt * 16 + half * 8 + j) * DINNER + h * 64;
#pragma unroll
    for (int ni = 0; ni < 4; ++ni) Ybuf[orow + ni * 16 + nn] = y[ni][j];
  }

  // ---- chunk states: states[p,n] = sum_s xdt[s,p]*decay[s]*B[s,n] ----
  float* sb = states + (((size_t)(b * NCHUNK + c) * HSSM + h) * HEADDIM) * DSTATE;
#pragma unroll
  for (int tt = 0; tt < 2; ++tt) {
    int t = wave * 2 + tt;
    int pt = t >> 3, nt = t & 7;
    v8f sacc = zero8();
    for (int k0 = 0; k0 < CHUNK; k0 += 32) {
      v16bf af;
      int m = pt * 16 + nn;
#pragma unroll
      for (int i = 0; i < 8; ++i) {
        int s = k0 + half * 16 + i;
        af[i] = (__bf16)((float)xdt[s][m] * dst_s[s]);
      }
#pragma unroll
      for (int i = 0; i < 8; ++i) {
        int s = k0 + half * 16 + 8 + i;
        af[8 + i] = (__bf16)((float)xdt[s][m] * dst_s[s]);
      }
      v16bf bf = load_b_g(xbc + (rowbase + k0) * CONVDIM + DINNER + nt * 16,
                          CONVDIM, lane);
      sacc = wmma_bf16(af, bf, sacc);
    }
#pragma unroll
    for (int j = 0; j < 8; ++j)
      sb[(size_t)(pt * 16 + half * 8 + j) * DSTATE + nt * 16 + nn] = sacc[j];
  }
}

// ---------------------------------------------------------------------------
// Inter-chunk recurrence: sprev[c] = S;  S = S*exp(Alast[c]) + states[c]
// ---------------------------------------------------------------------------
__global__ __launch_bounds__(256) void scan_states(const float* __restrict__ states,
                                                   const float* __restrict__ Alast,
                                                   float* __restrict__ sprev) {
  int t = blockIdx.x * 256 + threadIdx.x;
  if (t >= BB * HSSM * HEADDIM * DSTATE) return;
  int n = t & 127;
  int p = (t >> 7) & 63;
  int h = (t >> 13) & 31;
  int b = t >> 18;
  float S = 0.f;
  for (int c = 0; c < NCHUNK; ++c) {
    size_t idx = ((((size_t)b * NCHUNK + c) * HSSM + h) * HEADDIM + p) * DSTATE + n;
    sprev[idx] = S;
    S = S * __expf(Alast[(b * HSSM + h) * NCHUNK + c]) + states[idx];
  }
}

// ---------------------------------------------------------------------------
// Y_off = exp(Acs[l]) * (C[l] . sprev^T)  plus D*x, accumulated into Ybuf.
// ---------------------------------------------------------------------------
__global__ __launch_bounds__(512) void ssd_off(const float* __restrict__ xbc,
                                               const float* __restrict__ sprev,
                                               const float* __restrict__ Acs,
                                               const float* __restrict__ Dv,
                                               float* __restrict__ Ybuf) {
  int blk = blockIdx.x;
  int h = blk & 31;
  int c = (blk >> 5) & 7;
  int b = blk >> 8;
  int tid = threadIdx.x, lane = tid & 31, wave = tid >> 5;
  int half = lane >> 4, nn = lane & 15;
  __shared__ float acs_s[CHUNK];
  const float* acsrow = Acs + ((size_t)((b * HSSM + h) * NCHUNK) + c) * CHUNK;
  for (int i = tid; i < CHUNK; i += 512) acs_s[i] = acsrow[i];
  __syncthreads();
  size_t rowbase = (size_t)b * LL + c * CHUNK;
  int lt = wave;
  const float* Cbase = xbc + (rowbase + lt * 16) * CONVDIM + DINNER + DSTATE;
  const float* sb = sprev + (((size_t)(b * NCHUNK + c) * HSSM + h) * HEADDIM) * DSTATE;
  v16bf ca[4];
#pragma unroll
  for (int kk = 0; kk < 4; ++kk)
    ca[kk] = load_a_g(Cbase + kk * 32, CONVDIM, lane, 1.f);
  v8f y[4];
#pragma unroll
  for (int pt = 0; pt < 4; ++pt) y[pt] = zero8();
#pragma unroll
  for (int kk = 0; kk < 4; ++kk) {
#pragma unroll
    for (int pt = 0; pt < 4; ++pt) {
      v16bf bf = load_bT_g(sb + (size_t)pt * 16 * DSTATE + kk * 32, DSTATE, lane);
      y[pt] = wmma_bf16(ca[kk], bf, y[pt]);
    }
  }
  float Dh = Dv[h];
#pragma unroll
  for (int j = 0; j < 8; ++j) {
    int lrow = lt * 16 + half * 8 + j;
    float eA = __expf(acs_s[lrow]);
    size_t grow = rowbase + lrow;
#pragma unroll
    for (int pt = 0; pt < 4; ++pt) {
      int col = pt * 16 + nn;
      float xh = xbc[grow * CONVDIM + h * 64 + col];
      Ybuf[grow * DINNER + h * 64 + col] += y[pt][j] * eA + Dh * xh;
    }
  }
}

// ---------------------------------------------------------------------------
// Gated RMS norm: out = rmsnorm(Y * silu(z)) * nw,  width 2048.
// ---------------------------------------------------------------------------
__global__ __launch_bounds__(256) void gatednorm(const float* __restrict__ Y,
                                                 const float* __restrict__ zx,
                                                 const float* __restrict__ nw,
                                                 float* __restrict__ out) {
  int row = blockIdx.x;
  float tv[8];
  float ss = 0.f;
#pragma unroll
  for (int i = 0; i < 8; ++i) {
    int cc = threadIdx.x + (i << 8);
    float zv = zx[(size_t)row * DINPROJ + cc];
    float t = Y[(size_t)row * DINNER + cc] * (zv / (1.f + __expf(-zv)));
    tv[i] = t;
    ss += t * t;
  }
#pragma unroll
  for (int off = 16; off; off >>= 1) ss += __shfl_xor(ss, off, 32);
  __shared__ float red[8];
  if ((threadIdx.x & 31) == 0) red[threadIdx.x >> 5] = ss;
  __syncthreads();
  float tot = 0.f;
#pragma unroll
  for (int i = 0; i < 8; ++i) tot += red[i];
  float r = rsqrtf(tot / (float)DINNER + 1e-5f);
#pragma unroll
  for (int i = 0; i < 8; ++i) {
    int cc = threadIdx.x + (i << 8);
    out[(size_t)row * DINNER + cc] = tv[i] * r * nw[cc];
  }
}

// ---------------------------------------------------------------------------
// SwiGLU: out[row, c] = y[row, c] * silu(y[row, 2048 + c])
// ---------------------------------------------------------------------------
__global__ __launch_bounds__(256) void swiglu(const float* __restrict__ y,
                                              float* __restrict__ o) {
  int idx = blockIdx.x * 256 + threadIdx.x;
  const int total = MM * DFF;
  if (idx >= total) return;
  int col = idx & (DFF - 1);
  int row = idx >> 11;
  float a = y[(size_t)row * (2 * DFF) + col];
  float g = y[(size_t)row * (2 * DFF) + DFF + col];
  o[idx] = a * (g / (1.f + __expf(-g)));
}

// ---------------------------------------------------------------------------
// Orchestration. Workspace layout (floats), total ~57.0M floats (~228 MB).
// ---------------------------------------------------------------------------
extern "C" void kernel_launch(void* const* d_in, const int* in_sizes, int n_in,
                              void* d_out, int out_size, void* d_ws, size_t ws_size,
                              hipStream_t stream) {
  (void)in_sizes; (void)n_in; (void)out_size; (void)ws_size;
  const float* x_in        = (const float*)d_in[0];
  const float* attn_norm_w = (const float*)d_in[1];
  const float* Wqkv_w      = (const float*)d_in[2];
  const float* Wqkv_b      = (const float*)d_in[3];
  const float* attn_out_w  = (const float*)d_in[4];
  const float* attn_out_b  = (const float*)d_in[5];
  const float* normf_w     = (const float*)d_in[6];
  const float* m_in_w      = (const float*)d_in[7];
  const float* m_conv_w    = (const float*)d_in[8];
  const float* m_conv_b    = (const float*)d_in[9];
  const float* m_dt_bias   = (const float*)d_in[10];
  const float* m_A_log     = (const float*)d_in[11];
  const float* m_D         = (const float*)d_in[12];
  const float* m_norm_w    = (const float*)d_in[13];
  const float* m_out_w     = (const float*)d_in[14];
  const float* fc1_w       = (const float*)d_in[15];
  const float* fc2_w       = (const float*)d_in[16];
  const float* final_norm_w= (const float*)d_in[17];
  float* out = (float*)d_out;

  float* ws     = (float*)d_ws;
  float* slabA  = ws;                                   // M*4384 (qkv / zxbcdt / fc1)
  float* slabB  = slabA + (size_t)MM * DINPROJ;         // M*2304 (xBC / gated / ffin)
  float* slabC  = slabB + (size_t)MM * CONVDIM;         // M*2048 (Ybuf)
  float* slabD  = slabC + (size_t)MM * DINNER;          // M*1024 (h / ff)
  float* xb0    = slabD + (size_t)MM * DMODEL;          // M*1024
  float* xb1    = xb0 + (size_t)MM * DMODEL;            // M*1024
  float* states = xb1 + (size_t)MM * DMODEL;            // 4.19M
  float* sprev  = states + (size_t)BB * NCHUNK * HSSM * HEADDIM * DSTATE;
  float* dtbuf  = sprev + (size_t)BB * NCHUNK * HSSM * HEADDIM * DSTATE;  // M*32
  float* acs    = dtbuf + (size_t)MM * HSSM;            // B*H*NC*256
  float* alast  = acs + (size_t)BB * HSSM * NCHUNK * CHUNK;  // B*H*NC

  auto ggrid = [](int N) { return dim3(MM / 128, (N + 63) / 64); };

  // h = rmsnorm(x)
  addnorm<<<MM, 256, 0, stream>>>(x_in, nullptr, 0.f, attn_norm_w, slabD);
  // qkv
  gemm_xwt<<<ggrid(3 * DMODEL), 256, 0, stream>>>(slabD, Wqkv_w, Wqkv_b, slabA,
                                                  MM, 3 * DMODEL, DMODEL);
  // attention
  attn_fwd<<<dim3(LL / 16, BB * NHEAD), 32, 0, stream>>>(slabA, xb0);
  // out proj
  gemm_xwt<<<ggrid(DMODEL), 256, 0, stream>>>(xb0, attn_out_w, attn_out_b, xb1,
                                              MM, DMODEL, DMODEL);
  // x = rmsnorm(o + x)
  addnorm<<<MM, 256, 0, stream>>>(xb1, x_in, 1.0f, normf_w, xb0);

  float* cur = xb0;
  float* nxt = xb1;
  for (int i = 0; i < 4; ++i) {
    gemm_xwt<<<ggrid(DINPROJ), 256, 0, stream>>>(
        cur, m_in_w + (size_t)i * DINPROJ * DMODEL, nullptr, slabA,
        MM, DINPROJ, DMODEL);
    convsilu<<<(MM * CONVDIM + 255) / 256, 256, 0, stream>>>(
        slabA, m_conv_w + (size_t)i * CONVDIM * DCONV, m_conv_b + (size_t)i * CONVDIM,
        slabB);
    dtprep<<<2, 256, 0, stream>>>(slabA, m_dt_bias + i * HSSM, m_A_log + i * HSSM,
                                  dtbuf, acs, alast);
    ssd_local<<<BB * NCHUNK * HSSM, 512, 0, stream>>>(slabB, dtbuf, acs, alast,
                                                      slabC, states);
    scan_states<<<(BB * HSSM * HEADDIM * DSTATE + 255) / 256, 256, 0, stream>>>(
        states, alast, sprev);
    ssd_off<<<BB * NCHUNK * HSSM, 512, 0, stream>>>(slabB, sprev, acs,
                                                    m_D + i * HSSM, slabC);
    gatednorm<<<MM, 256, 0, stream>>>(slabC, slabA, m_norm_w + (size_t)i * DINNER,
                                      slabB);
    gemm_xwt<<<ggrid(DMODEL), 256, 0, stream>>>(
        slabB, m_out_w + (size_t)i * DMODEL * DINNER, nullptr, nxt,
        MM, DMODEL, DINNER);
    float* t = cur; cur = nxt; nxt = t;
  }

  // MLP
  gemm_xwt<<<ggrid(2 * DFF), 256, 0, stream>>>(cur, fc1_w, nullptr, slabA,
                                               MM, 2 * DFF, DMODEL);
  swiglu<<<(MM * DFF + 255) / 256, 256, 0, stream>>>(slabA, slabB);
  gemm_xwt<<<ggrid(DMODEL), 256, 0, stream>>>(slabB, fc2_w, nullptr, slabD,
                                              MM, DMODEL, DFF);
  // out = rmsnorm(ff + 1.68*x)
  addnorm<<<MM, 256, 0, stream>>>(slabD, cur, 1.68f, final_norm_w, out);
}
